// MultiHeadAttention_90469191123419
// MI455X (gfx1250) — matrix operand order repaired and checked by reading, not compiled, not run on hardware
//
#include <hip/hip_runtime.h>
#include <hip/hip_bf16.h>

typedef __attribute__((ext_vector_type(16))) _Float16 v16h;
typedef __attribute__((ext_vector_type(8)))  float    v8f;
typedef int v4i __attribute__((vector_size(4 * sizeof(int))));

#define BB 2
#define SS 2048
#define DD 1024
#define HH 16
#define DKK 64
#define MM (BB*SS)   // 4096

union AFrag { v16h v; unsigned u[8]; };
union BFrag { v16h v; unsigned u[8]; };

#if defined(__AMDGCN__) && __has_builtin(__builtin_amdgcn_global_load_async_to_lds_b128) && __has_builtin(__builtin_amdgcn_s_wait_asynccnt)
#define USE_ASYNC_LDS 1
#else
#define USE_ASYNC_LDS 0
#endif

// ---------------------------------------------------------------------------
// lengths[b] = count_nonzero(mask[b,:])
// ---------------------------------------------------------------------------
__global__ void mha_lengths_kernel(const int* __restrict__ mask, int* __restrict__ lengths) {
    __shared__ int red[256];
    int b = blockIdx.x;
    int cnt = 0;
    for (int i = threadIdx.x; i < SS; i += 256)
        cnt += (mask[b * SS + i] != 0) ? 1 : 0;
    red[threadIdx.x] = cnt;
    __syncthreads();
    for (int s = 128; s > 0; s >>= 1) {
        if (threadIdx.x < s) red[threadIdx.x] += red[threadIdx.x + s];
        __syncthreads();
    }
    if (threadIdx.x == 0) lengths[b] = red[0];
}

// ---------------------------------------------------------------------------
// f32 -> f16 convert (n multiple of 4)
// ---------------------------------------------------------------------------
__global__ void mha_cvt_f16_kernel(const float* __restrict__ src, _Float16* __restrict__ dst, int n) {
    int i = (blockIdx.x * blockDim.x + threadIdx.x) * 4;
    if (i < n) {
        float4 f = *(const float4*)&src[i];
        union { _Float16 h[4]; unsigned u[2]; } pk;
        pk.h[0] = (_Float16)f.x; pk.h[1] = (_Float16)f.y;
        pk.h[2] = (_Float16)f.z; pk.h[3] = (_Float16)f.w;
        *(unsigned*)&dst[i]     = pk.u[0];
        *(unsigned*)&dst[i + 2] = pk.u[1];
    }
}

// ---------------------------------------------------------------------------
// C[M,N] = A[M,K] @ W[N,K]^T + bias   (nn.Linear semantics)
// Block: 256 thr = 8 waves (4 m-waves x 2 n-waves); block tile 128(M) x 64(N);
// each wave computes a 32x32 C tile (4 accumulators); K-step 64;
// double-buffered LDS staging via GLOBAL_LOAD_ASYNC_TO_LDS_B128 (ASYNCcnt)
// when the toolchain exposes the builtin, sync uint4 copies otherwise.
// ---------------------------------------------------------------------------
template<bool HALF_OUT>
__global__ void __launch_bounds__(256) mha_gemm_wmma_kernel(
    const _Float16* __restrict__ A, const _Float16* __restrict__ W,
    const float* __restrict__ bias, void* __restrict__ Cout,
    int Mtot, int Ntot, int Ktot)
{
    __shared__ __align__(16) _Float16 As[2][128][72];  // 128 x 64 halves (+pad)
    __shared__ __align__(16) _Float16 Ws[2][64][72];   // 64 x 64 halves (+pad)

    const int tid  = threadIdx.x;
    const int wave = tid >> 5;
    const int lane = tid & 31;
    const int hg   = (lane >> 4) & 1;   // half-group
    const int ln   = lane & 15;
    const int wm   = wave & 3;          // 4 m-waves
    const int wn   = wave >> 2;         // 2 n-waves
    const int m0 = blockIdx.x * 128;
    const int n0 = blockIdx.y * 64;

    // staging decomposition:
    //   A tile: 128 rows x 8 chunks(16B) -> 4 chunks/thread
    const int arow = tid >> 1, asel = (tid & 1) * 32;
    //   W tile: 64 rows x 8 chunks(16B)  -> 2 chunks/thread
    const int brow = tid >> 2, bsel = (tid & 3) * 16;

    auto stage = [&](int buf, int k0) {
        #pragma unroll
        for (int c = 0; c < 4; ++c) {
            const _Float16* g = &A[(size_t)(m0 + arow) * Ktot + k0 + asel + c * 8];
            _Float16* l = &As[buf][arow][asel + c * 8];
#if USE_ASYNC_LDS
            __builtin_amdgcn_global_load_async_to_lds_b128(
                (__attribute__((address_space(1))) v4i*)g,
                (__attribute__((address_space(3))) v4i*)l, 0, 0);
#else
            *(uint4*)l = *(const uint4*)g;
#endif
        }
        #pragma unroll
        for (int c = 0; c < 2; ++c) {
            const _Float16* g = &W[(size_t)(n0 + brow) * Ktot + k0 + bsel + c * 8];
            _Float16* l = &Ws[buf][brow][bsel + c * 8];
#if USE_ASYNC_LDS
            __builtin_amdgcn_global_load_async_to_lds_b128(
                (__attribute__((address_space(1))) v4i*)g,
                (__attribute__((address_space(3))) v4i*)l, 0, 0);
#else
            *(uint4*)l = *(const uint4*)g;
#endif
        }
    };

    v8f acc[2][2];
    #pragma unroll
    for (int a = 0; a < 2; ++a)
        #pragma unroll
        for (int b = 0; b < 2; ++b)
            #pragma unroll
            for (int i = 0; i < 8; ++i) acc[a][b][i] = 0.0f;

    stage(0, 0);

    for (int k0 = 0; k0 < Ktot; k0 += 64) {
        const int buf = (k0 >> 6) & 1;
        if (k0 + 64 < Ktot) {
            stage(buf ^ 1, k0 + 64);       // overlap next-tile DMA with compute
#if USE_ASYNC_LDS
            __builtin_amdgcn_s_wait_asynccnt(6);   // 6 in flight for next tile
#endif
        } else {
#if USE_ASYNC_LDS
            __builtin_amdgcn_s_wait_asynccnt(0);
#endif
        }
        __syncthreads();

        // fragments for this wave's 32x32 tile (2 m-subtiles x 2 n-subtiles,
        // K split into two 32-wide WMMAs)
        AFrag af[2][2];
        #pragma unroll
        for (int ms = 0; ms < 2; ++ms)
            #pragma unroll
            for (int kw = 0; kw < 2; ++kw)
                #pragma unroll
                for (int v = 0; v < 8; ++v) {
                    int kk = 2 * v + (hg ? 16 : 0) + kw * 32;
                    af[ms][kw].u[v] = *(const unsigned*)&As[buf][wm * 32 + ms * 16 + ln][kk];
                }
        BFrag bf[2][2];
        #pragma unroll
        for (int ns = 0; ns < 2; ++ns)
            #pragma unroll
            for (int kw = 0; kw < 2; ++kw)
                #pragma unroll
                for (int v = 0; v < 8; ++v) {
                    int kk = 2 * v + (hg ? 16 : 0) + kw * 32;
                    bf[ns][kw].u[v] = *(const unsigned*)&Ws[buf][wn * 32 + ns * 16 + ln][kk];
                }
        #pragma unroll
        for (int ms = 0; ms < 2; ++ms)
            #pragma unroll
            for (int ns = 0; ns < 2; ++ns) {
                acc[ms][ns] = __builtin_amdgcn_wmma_f32_16x16x32_f16(
                    false, af[ms][0].v, false, bf[ns][0].v, (short)0, acc[ms][ns], false, false);
                acc[ms][ns] = __builtin_amdgcn_wmma_f32_16x16x32_f16(
                    false, af[ms][1].v, false, bf[ns][1].v, (short)0, acc[ms][ns], false, false);
            }
        __syncthreads();
    }

    #pragma unroll
    for (int ns = 0; ns < 2; ++ns) {
        const int col = n0 + wn * 32 + ns * 16 + ln;
        const float bv = bias[col];
        #pragma unroll
        for (int ms = 0; ms < 2; ++ms)
            #pragma unroll
            for (int i = 0; i < 8; ++i) {
                int r = m0 + wm * 32 + ms * 16 + i + (hg ? 8 : 0);
                float val = acc[ms][ns][i] + bv;
                if (HALF_OUT)
                    ((_Float16*)Cout)[(size_t)r * Ntot + col] = (_Float16)val;
                else
                    ((float*)Cout)[(size_t)r * Ntot + col] = val;
            }
    }
}

// ---------------------------------------------------------------------------
// Flash attention: out(concat f16) = softmax(Q K^T / sqrt(DK) + prefix mask) V
// grid: (S/128, H, B); 8 waves/block, 1 wave per 16-query tile, shared (b,h).
// ---------------------------------------------------------------------------
__global__ void __launch_bounds__(256) mha_attn_wmma_kernel(
    const _Float16* __restrict__ Q, const _Float16* __restrict__ Kc,
    const _Float16* __restrict__ V, const int* __restrict__ lengths,
    _Float16* __restrict__ Oc)
{
    __shared__ __align__(16) _Float16 Kt[32][72];    // 32 keys x 64 d (padded)
    __shared__ __align__(16) _Float16 VtT[64][34];   // transposed: 64 d x 32 keys
    __shared__ __align__(16) _Float16 Pl[8][16][40]; // per-wave P transpose buffer

    const int tid  = threadIdx.x;
    const int wave = tid >> 5;
    const int lane = tid & 31;
    const int hg   = (lane >> 4) & 1;
    const int ln   = lane & 15;

    const int b  = blockIdx.z;
    const int h  = blockIdx.y;
    const int q0 = blockIdx.x * 128 + wave * 16;
    const int len = lengths[b];

    // Q fragments (A-layout, K-dim = d, split d into two 32-chunks), kept in regs
    AFrag qa[2];
    #pragma unroll
    for (int dw = 0; dw < 2; ++dw)
        #pragma unroll
        for (int v = 0; v < 8; ++v) {
            int kk = 2 * v + (hg ? 16 : 0) + dw * 32;
            qa[dw].u[v] = *(const unsigned*)&Q[((size_t)(b * SS + q0 + ln)) * DD + h * DKK + kk];
        }

    float mrow[8], lrow[8];
    v8f oacc[4];
    #pragma unroll
    for (int i = 0; i < 8; ++i) { mrow[i] = -3.0e38f; lrow[i] = 0.0f; }
    #pragma unroll
    for (int j = 0; j < 4; ++j)
        #pragma unroll
        for (int i = 0; i < 8; ++i) oacc[j][i] = 0.0f;

    const int kkey = tid >> 3, kchk = tid & 7; // staging: 32 keys x 8 chunks of 8 halves
    int kb_end = (len + 31) / 32;
    if (kb_end > SS / 32) kb_end = SS / 32;

    for (int kb = 0; kb < kb_end; ++kb) {
        const int kbase = kb * 32;
        // stage K tile row-major (b128)
        *(uint4*)&Kt[kkey][kchk * 8] =
            *(const uint4*)&Kc[((size_t)(b * SS + kbase + kkey)) * DD + h * DKK + kchk * 8];
        // stage V transposed (so P*V B-frag K-pairs are contiguous)
        {
            union { uint4 q; _Float16 hv[8]; } t;
            t.q = *(const uint4*)&V[((size_t)(b * SS + kbase + kkey)) * DD + h * DKK + kchk * 8];
            #pragma unroll
            for (int j = 0; j < 8; ++j) VtT[kchk * 8 + j][kkey] = t.hv[j];
        }
        // prefetch next K tile (gfx1250 global_prefetch_b8)
        if (kb + 1 < kb_end)
            __builtin_prefetch(&Kc[((size_t)(b * SS + kbase + 32 + kkey)) * DD + h * DKK + kchk * 8], 0, 0);
        __syncthreads();

        // scores: two 16-key subtiles, K-dim = d (64) split into two WMMAs
        v8f s[2];
        #pragma unroll
        for (int nt = 0; nt < 2; ++nt) {
            #pragma unroll
            for (int i = 0; i < 8; ++i) s[nt][i] = 0.0f;
            #pragma unroll
            for (int dw = 0; dw < 2; ++dw) {
                BFrag kf;
                #pragma unroll
                for (int v = 0; v < 8; ++v) {
                    int kk = 2 * v + (hg ? 16 : 0) + dw * 32;
                    kf.u[v] = *(const unsigned*)&Kt[nt * 16 + ln][kk];
                }
                s[nt] = __builtin_amdgcn_wmma_f32_16x16x32_f16(
                    false, qa[dw].v, false, kf.v, (short)0, s[nt], false, false);
            }
            // scale + prefix-length mask (column uniform per lane)
            const bool valid = (kbase + nt * 16 + ln) < len;
            #pragma unroll
            for (int i = 0; i < 8; ++i)
                s[nt][i] = valid ? s[nt][i] * 0.125f : -3.0e38f;
        }

        // online softmax per row (rows of a half-group live across its 16 lanes)
        #pragma unroll
        for (int i = 0; i < 8; ++i) {
            float t = fmaxf(s[0][i], s[1][i]);
            t = fmaxf(t, __shfl_xor(t, 1, 32));
            t = fmaxf(t, __shfl_xor(t, 2, 32));
            t = fmaxf(t, __shfl_xor(t, 4, 32));
            t = fmaxf(t, __shfl_xor(t, 8, 32));
            float mn = fmaxf(mrow[i], t);
            float alpha = __expf(mrow[i] - mn);
            mrow[i] = mn;
            float p0 = __expf(s[0][i] - mn);
            float p1 = __expf(s[1][i] - mn);
            float ps = p0 + p1;
            ps += __shfl_xor(ps, 1, 32);
            ps += __shfl_xor(ps, 2, 32);
            ps += __shfl_xor(ps, 4, 32);
            ps += __shfl_xor(ps, 8, 32);
            lrow[i] = lrow[i] * alpha + ps;
            #pragma unroll
            for (int j = 0; j < 4; ++j) oacc[j][i] *= alpha;
            // write P (C-layout) into per-wave LDS for A-layout reload
            Pl[wave][hg * 8 + i][ln]      = (_Float16)p0;
            Pl[wave][hg * 8 + i][16 + ln] = (_Float16)p1;
        }

        // P as A-fragment (16x32, K-dim = keys); DS ops are in-order per wave
        AFrag pa;
        #pragma unroll
        for (int v = 0; v < 8; ++v) {
            int kk = 2 * v + (hg ? 16 : 0);
            pa.u[v] = *(const unsigned*)&Pl[wave][ln][kk];
        }
        // O += P @ V : 4 output column subtiles
        #pragma unroll
        for (int j = 0; j < 4; ++j) {
            BFrag vb;
            #pragma unroll
            for (int v = 0; v < 8; ++v) {
                int kk = 2 * v + (hg ? 16 : 0);
                vb.u[v] = *(const unsigned*)&VtT[j * 16 + ln][kk];
            }
            oacc[j] = __builtin_amdgcn_wmma_f32_16x16x32_f16(
                false, pa.v, false, vb.v, (short)0, oacc[j], false, false);
        }
        __syncthreads();
    }

    // epilogue: divide by l, emit concat (f16) for the output projection
    #pragma unroll
    for (int i = 0; i < 8; ++i) {
        float inv = (lrow[i] != 0.0f) ? (1.0f / lrow[i]) : 0.0f;
        int row = q0 + hg * 8 + i;
        #pragma unroll
        for (int j = 0; j < 4; ++j) {
            int col = h * DKK + j * 16 + ln;
            Oc[((size_t)(b * SS + row)) * DD + col] = (_Float16)(oacc[j][i] * inv);
        }
    }
}

// ---------------------------------------------------------------------------
extern "C" void kernel_launch(void* const* d_in, const int* in_sizes, int n_in,
                              void* d_out, int out_size, void* d_ws, size_t ws_size,
                              hipStream_t stream) {
    const float* x    = (const float*)d_in[0];
    const int*   mask = (const int*)d_in[1];
    const float* Wq_w = (const float*)d_in[2];
    const float* Wq_b = (const float*)d_in[3];
    const float* Wk_w = (const float*)d_in[4];
    const float* Wk_b = (const float*)d_in[5];
    const float* Wv_w = (const float*)d_in[6];
    const float* Wv_b = (const float*)d_in[7];
    const float* Wo_w = (const float*)d_in[8];
    const float* Wo_b = (const float*)d_in[9];
    float* out = (float*)d_out;

    const size_t MD  = (size_t)MM * DD;  // 4096*1024 elements
    const size_t DDs = (size_t)DD * DD;  // 1024*1024 elements

    char* base = (char*)d_ws;
    _Float16* xh  = (_Float16*)(base);
    _Float16* wqh = (_Float16*)(base + 2 * (MD));
    _Float16* wkh = (_Float16*)(base + 2 * (MD + 1 * DDs));
    _Float16* wvh = (_Float16*)(base + 2 * (MD + 2 * DDs));
    _Float16* woh = (_Float16*)(base + 2 * (MD + 3 * DDs));
    _Float16* qh  = (_Float16*)(base + 2 * (MD + 4 * DDs));
    _Float16* kh  = (_Float16*)(base + 2 * (2 * MD + 4 * DDs));
    _Float16* vh  = (_Float16*)(base + 2 * (3 * MD + 4 * DDs));
    _Float16* ch  = (_Float16*)(base + 2 * (4 * MD + 4 * DDs));
    int* lengths  = (int*)(base + 2 * (5 * MD + 4 * DDs));

    // 1) prefix lengths
    mha_lengths_kernel<<<BB, 256, 0, stream>>>(mask, lengths);

    // 2) f32 -> f16 conversions
    mha_cvt_f16_kernel<<<(int)(MD / 1024), 256, 0, stream>>>(x, xh, (int)MD);
    mha_cvt_f16_kernel<<<(int)(DDs / 1024), 256, 0, stream>>>(Wq_w, wqh, (int)DDs);
    mha_cvt_f16_kernel<<<(int)(DDs / 1024), 256, 0, stream>>>(Wk_w, wkh, (int)DDs);
    mha_cvt_f16_kernel<<<(int)(DDs / 1024), 256, 0, stream>>>(Wv_w, wvh, (int)DDs);
    mha_cvt_f16_kernel<<<(int)(DDs / 1024), 256, 0, stream>>>(Wo_w, woh, (int)DDs);

    // 3) Q/K/V projections (f16 out)
    dim3 ggrid(MM / 128, DD / 64);
    mha_gemm_wmma_kernel<true><<<ggrid, 256, 0, stream>>>(xh, wqh, Wq_b, qh, MM, DD, DD);
    mha_gemm_wmma_kernel<true><<<ggrid, 256, 0, stream>>>(xh, wkh, Wk_b, kh, MM, DD, DD);
    mha_gemm_wmma_kernel<true><<<ggrid, 256, 0, stream>>>(xh, wvh, Wv_b, vh, MM, DD, DD);

    // 4) flash attention -> concat (f16)
    dim3 agrid(SS / 128, HH, BB);
    mha_attn_wmma_kernel<<<agrid, 256, 0, stream>>>(qh, kh, vh, lengths, ch);

    // 5) output projection (f32 out)
    mha_gemm_wmma_kernel<false><<<ggrid, 256, 0, stream>>>(ch, woh, Wo_b, out, MM, DD, DD);
}
